// MoEMLPWithEinops_79688823210284
// MI455X (gfx1250) — matrix operand order repaired and checked by reading, not compiled, not run on hardware
//
#include <hip/hip_runtime.h>
#include <math.h>

#define B_     2
#define S_     2048
#define D_     1024
#define M_     2048
#define E_     8
#define TOKENS (B_ * S_)      // 4096
#define KC     32             // WMMA K chunk (bf16 k=32)
#define LDP    40             // LDS row stride in elements (80B: 16B-aligned, bank-spread)

typedef __bf16 bf16;
typedef bf16  v16bf __attribute__((ext_vector_type(16)));
typedef bf16  v8bf  __attribute__((ext_vector_type(8)));
typedef bf16  v4bf  __attribute__((ext_vector_type(4)));
typedef float v8f   __attribute__((ext_vector_type(8)));

// ---------------- CDNA5 async global->LDS DMA (tracked by ASYNCcnt) ----------------
__device__ inline void async_b128(unsigned lds_off, const bf16* gaddr) {
    asm volatile("global_load_async_to_lds_b128 %0, %1, off"
                 :: "v"(lds_off), "v"(gaddr) : "memory");
}
__device__ inline void wait_async0() { asm volatile("s_wait_asynccnt 0x0" ::: "memory"); }
__device__ inline void wait_async2() { asm volatile("s_wait_asynccnt 0x2" ::: "memory"); }
__device__ inline void wait_async3() { asm volatile("s_wait_asynccnt 0x3" ::: "memory"); }
__device__ inline unsigned lds_off_of(const void* p) {
    return (unsigned)(size_t)p;   // low 32 bits of shared-aperture address = LDS offset
}

// ---------------- fp32 -> bf16 cast (4-wide, for residual) ----------------
__global__ void cast_f32_bf16_x4(const float* __restrict__ src,
                                 bf16* __restrict__ dst, int n4) {
    int i = blockIdx.x * blockDim.x + threadIdx.x;
    if (i < n4) {
        float4 v = ((const float4*)src)[i];
        v4bf r;
        r[0] = (bf16)v.x; r[1] = (bf16)v.y; r[2] = (bf16)v.z; r[3] = (bf16)v.w;
        ((v4bf*)dst)[i] = r;
    }
}

// ---------------- fp32 [R][C] -> bf16 transposed [C][R] (per expert) ----------------
__global__ __launch_bounds__(256)
void transpose_cast_kernel(const float* __restrict__ src, bf16* __restrict__ dst,
                           int R, int C) {
    __shared__ bf16 T[32][33];
    int e  = blockIdx.z;
    int r0 = blockIdx.y * 32;
    int c0 = blockIdx.x * 32;
    int tid = threadIdx.x;
    const float* s = src + (size_t)e * R * C;
    bf16*        d = dst + (size_t)e * R * C;
    {   // load 32x32 fp32 tile, coalesced along C
        int cl = tid & 31, rq = tid >> 5;
#pragma unroll
        for (int j = 0; j < 4; j++)
            T[rq + 8 * j][cl] = (bf16)s[(size_t)(r0 + rq + 8 * j) * C + c0 + cl];
    }
    __syncthreads();
    {   // store transposed, coalesced along R, 4-wide
        int rl = tid >> 3, dq = (tid & 7) * 4;
        v4bf v;
#pragma unroll
        for (int j = 0; j < 4; j++) v[j] = T[dq + j][rl];
        *(v4bf*)&d[(size_t)(c0 + rl) * R + r0 + dq] = v;
    }
}

// ---------------- router: softmax + top-2 + expert lists ----------------
__global__ void router_kernel(const float* __restrict__ X,
                              const float* __restrict__ Wr,
                              int* __restrict__ counts,
                              int* __restrict__ list_tok,
                              float* __restrict__ list_w) {
    int tok  = blockIdx.x;
    int lane = threadIdx.x;          // one wave32
    float acc[E_];
#pragma unroll
    for (int e = 0; e < E_; e++) acc[e] = 0.f;
    const float* x = X + (size_t)tok * D_;
    for (int d = lane; d < D_; d += 32) {
        float xv = x[d];
#pragma unroll
        for (int e = 0; e < E_; e++) acc[e] += xv * Wr[d * E_ + e];
    }
#pragma unroll
    for (int off = 16; off > 0; off >>= 1) {
#pragma unroll
        for (int e = 0; e < E_; e++) acc[e] += __shfl_xor(acc[e], off, 32);
    }
    if (lane == 0) {
        float mx = acc[0];
#pragma unroll
        for (int e = 1; e < E_; e++) mx = fmaxf(mx, acc[e]);
        float p[E_], sum = 0.f;
#pragma unroll
        for (int e = 0; e < E_; e++) { p[e] = __expf(acc[e] - mx); sum += p[e]; }
        int   i1 = 0, i2 = -1;
        float p1 = p[0], p2 = -1.f;
#pragma unroll
        for (int e = 1; e < E_; e++) {
            if (p[e] > p1) { p2 = p1; i2 = i1; p1 = p[e]; i1 = e; }
            else if (p[e] > p2) { p2 = p[e]; i2 = e; }
        }
        float pr1 = p1 / sum, pr2 = p2 / sum;
        float den = pr1 + pr2 + 1e-8f;
        int pos1 = atomicAdd(&counts[i1], 1);
        list_tok[i1 * TOKENS + pos1] = tok;
        list_w  [i1 * TOKENS + pos1] = pr1 / den;
        int pos2 = atomicAdd(&counts[i2], 1);
        list_tok[i2 * TOKENS + pos2] = tok;
        list_w  [i2 * TOKENS + pos2] = pr2 / den;
    }
}

__global__ void scan_kernel(const int* __restrict__ counts, int* __restrict__ offsets) {
    if (threadIdx.x == 0) {
        int s = 0;
        for (int e = 0; e < E_; e++) { offsets[e] = s; s += counts[e]; }
    }
}

// ---------------- fragment load: 16 contiguous bf16 (2x ds_load_b128) ----------------
__device__ inline v16bf ld16(const bf16* p) {
    v8bf lo = *(const v8bf*)p;
    v8bf hi = *(const v8bf*)(p + 8);
    return __builtin_shufflevector(lo, hi, 0, 1, 2, 3, 4, 5, 6, 7,
                                            8, 9, 10, 11, 12, 13, 14, 15);
}
// A-chunk permutation: 8-blocks stored in order [0..7, 16..23, 8..15, 24..31]
__device__ inline int a_perm_chunk(int c) { return (c == 1) ? 2 : (c == 2) ? 1 : c; }

// ---------------- fused gate/up GEMM + SwiGLU ----------------
// block = 256 threads (8 waves), tile = 64 tokens x 64 hidden cols
// wave (rt = w>>1, cp = w&1) owns a 16-row x 32-col slice: 4 WMMAs/K-step, shared A frag
// double-buffered async DMA: issue tile i+1, s_wait_asynccnt<=3 overlaps DMA with WMMA
__global__ __launch_bounds__(256)
void gateup_kernel(const bf16* __restrict__ Xb, const bf16* __restrict__ Wgt,
                   const bf16* __restrict__ Wut, const float* __restrict__ bg,
                   const float* __restrict__ bu,
                   const int* __restrict__ counts, const int* __restrict__ offsets,
                   const int* __restrict__ list_tok, bf16* __restrict__ H) {
    int e    = blockIdx.z;
    int cnt  = counts[e];
    int row0 = blockIdx.y * 64;
    if (row0 >= cnt) return;
    int n0 = blockIdx.x * 64;

    __shared__ bf16 As [2][64][LDP];   // A rows: [row][k], natural k order
    __shared__ bf16 Btg[2][64][LDP];   // gate weights (pre-transposed): [n][k]
    __shared__ bf16 Btu[2][64][LDP];   // up   weights (pre-transposed): [n][k]
    __shared__ __align__(16) bf16 Atr[8];  // DMA trash slot for invalid rows
    __shared__ int  toks[64];

    int tid  = threadIdx.x;
    int lane = tid & 31;
    int wave = tid >> 5;
    int wrow = (wave >> 1) * 16;   // 0,16,32,48
    int wcol = (wave & 1) * 32;    // 0,32

    if (tid < 64) {
        int p = row0 + tid;
        toks[tid] = (p < cnt) ? list_tok[e * TOKENS + p] : -1;
    }
    __syncthreads();

    // per-thread async-staging assignment (1 A chunk + 1 gate chunk + 1 up chunk)
    int r_a = tid >> 2, c_a = tid & 3;
    int pc8 = c_a * 8;
    int t_a = toks[r_a];
    bool va = (t_a >= 0);
    unsigned a_lds[2]  = { va ? lds_off_of(&As[0][r_a][pc8]) : lds_off_of(Atr),
                           va ? lds_off_of(&As[1][r_a][pc8]) : lds_off_of(Atr) };
    unsigned bg_lds[2] = { lds_off_of(&Btg[0][r_a][c_a * 8]),
                           lds_off_of(&Btg[1][r_a][c_a * 8]) };
    unsigned bu_lds[2] = { lds_off_of(&Btu[0][r_a][c_a * 8]),
                           lds_off_of(&Btu[1][r_a][c_a * 8]) };
    const bf16* a_src = Xb + (size_t)(va ? t_a : 0) * D_ + c_a * 8;
    const bf16* g_src = Wgt + (size_t)e * D_ * M_ + (size_t)(n0 + r_a) * D_ + c_a * 8;
    const bf16* u_src = Wut + (size_t)e * D_ * M_ + (size_t)(n0 + r_a) * D_ + c_a * 8;
    if (!va) {  // invalid rows stay zero in both buffers (their DMA targets the trash slot)
        v8bf z = {};
        *(v8bf*)&As[0][r_a][pc8] = z;
        *(v8bf*)&As[1][r_a][pc8] = z;
    }

    v8f cg0 = {}, cg1 = {}, cu0 = {}, cu1 = {};
    int hr   = lane & 15;          // fragment row/col within 16
    int half = (lane >> 4) * 16;   // 0 or 16: which contiguous half

    auto issue = [&](int buf, int k0) {
        async_b128(a_lds[buf],  a_src + k0);
        async_b128(bg_lds[buf], g_src + k0);
        async_b128(bu_lds[buf], u_src + k0);
    };

    constexpr int NIT = D_ / KC;
    issue(0, 0);
#pragma unroll 2
    for (int i = 0; i < NIT; i++) {
        int b = i & 1;
        if (i + 1 < NIT) { issue(b ^ 1, (i + 1) * KC); wait_async3(); }
        else             { wait_async0(); }
        __syncthreads();
        v16bf a   = ld16(&As [b][wrow + hr][half]);
        v16bf bgf0 = ld16(&Btg[b][wcol + hr][half]);
        v16bf bgf1 = ld16(&Btg[b][wcol + 16 + hr][half]);
        v16bf buf0 = ld16(&Btu[b][wcol + hr][half]);
        v16bf buf1 = ld16(&Btu[b][wcol + 16 + hr][half]);
        cg0 = __builtin_amdgcn_wmma_f32_16x16x32_bf16(false, a, false, bgf0, (short)0, cg0, false, false);
        cg1 = __builtin_amdgcn_wmma_f32_16x16x32_bf16(false, a, false, bgf1, (short)0, cg1, false, false);
        cu0 = __builtin_amdgcn_wmma_f32_16x16x32_bf16(false, a, false, buf0, (short)0, cu0, false, false);
        cu1 = __builtin_amdgcn_wmma_f32_16x16x32_bf16(false, a, false, buf1, (short)0, cu1, false, false);
        __syncthreads();
    }

    // epilogue: SiLU(gate)*up -> H (bf16, compacted rows)
    int n   = lane & 15;
    int mb  = (lane >> 4) * 8;
    int hbase = offsets[e] + row0;
#pragma unroll
    for (int f = 0; f < 2; f++) {
        int col = n0 + wcol + f * 16 + n;
        float bgv = bg[e * M_ + col];
        float buv = bu[e * M_ + col];
        const v8f& cgf = f ? cg1 : cg0;
        const v8f& cuf = f ? cu1 : cu0;
#pragma unroll
        for (int v = 0; v < 8; v++) {
            int rib = wrow + mb + v;
            if (row0 + rib < cnt) {
                float g = cgf[v] + bgv;
                float u = cuf[v] + buv;
                float s = g / (1.f + __expf(-g));
                H[(size_t)(hbase + rib) * M_ + col] = (bf16)(s * u);
            }
        }
    }
}

// ---------------- out projection GEMM + weighted scatter-add ----------------
// tile = 64 rows x 64 out-cols; wave owns 16x32 slice: 2 WMMAs/K-step; double-buffered DMA
__global__ __launch_bounds__(256)
void out_kernel(const bf16* __restrict__ H, const bf16* __restrict__ Wot,
                const float* __restrict__ bo, const int* __restrict__ counts,
                const int* __restrict__ offsets, const int* __restrict__ list_tok,
                const float* __restrict__ list_w, float* __restrict__ out) {
    int e    = blockIdx.z;
    int cnt  = counts[e];
    int row0 = blockIdx.y * 64;
    if (row0 >= cnt) return;
    int n0 = blockIdx.x * 64;

    __shared__ bf16  As[2][64][LDP];
    __shared__ bf16  Bt[2][64][LDP];   // W_out pre-transposed: [n=d][k=m]
    __shared__ __align__(16) bf16 Atr[8];
    __shared__ int   toks[64];
    __shared__ float rw[64];

    int tid  = threadIdx.x;
    int lane = tid & 31;
    int wave = tid >> 5;
    int wrow = (wave >> 1) * 16;
    int wcol = (wave & 1) * 32;

    if (tid < 64) {
        int p = row0 + tid;
        toks[tid] = (p < cnt) ? list_tok[e * TOKENS + p] : -1;
        rw[tid]   = (p < cnt) ? list_w[e * TOKENS + p] : 0.f;
    }
    __syncthreads();

    int hbase = offsets[e] + row0;
    int r_a = tid >> 2, c_a = tid & 3;
    int pc8 = c_a * 8;
    bool va = (row0 + r_a) < cnt;
    unsigned a_lds[2] = { va ? lds_off_of(&As[0][r_a][pc8]) : lds_off_of(Atr),
                          va ? lds_off_of(&As[1][r_a][pc8]) : lds_off_of(Atr) };
    unsigned b_lds[2] = { lds_off_of(&Bt[0][r_a][c_a * 8]),
                          lds_off_of(&Bt[1][r_a][c_a * 8]) };
    const bf16* a_src = H + (size_t)(hbase + (va ? r_a : 0)) * M_ + c_a * 8;
    const bf16* b_src = Wot + (size_t)e * M_ * D_ + (size_t)(n0 + r_a) * M_ + c_a * 8;
    if (!va) {
        v8bf z = {};
        *(v8bf*)&As[0][r_a][pc8] = z;
        *(v8bf*)&As[1][r_a][pc8] = z;
    }

    v8f c0 = {}, c1 = {};
    int hr   = lane & 15;
    int half = (lane >> 4) * 16;

    auto issue = [&](int buf, int k0) {
        async_b128(a_lds[buf], a_src + k0);
        async_b128(b_lds[buf], b_src + k0);
    };

    constexpr int NIT = M_ / KC;
    issue(0, 0);
#pragma unroll 2
    for (int i = 0; i < NIT; i++) {
        int b = i & 1;
        if (i + 1 < NIT) { issue(b ^ 1, (i + 1) * KC); wait_async2(); }
        else             { wait_async0(); }
        __syncthreads();
        v16bf a  = ld16(&As[b][wrow + hr][half]);
        v16bf b0 = ld16(&Bt[b][wcol + hr][half]);
        v16bf b1 = ld16(&Bt[b][wcol + 16 + hr][half]);
        c0 = __builtin_amdgcn_wmma_f32_16x16x32_bf16(false, a, false, b0, (short)0, c0, false, false);
        c1 = __builtin_amdgcn_wmma_f32_16x16x32_bf16(false, a, false, b1, (short)0, c1, false, false);
        __syncthreads();
    }

    int n  = lane & 15;
    int mb = (lane >> 4) * 8;
#pragma unroll
    for (int f = 0; f < 2; f++) {
        int col = n0 + wcol + f * 16 + n;
        float bov = bo[e * D_ + col];
        const v8f& cf = f ? c1 : c0;
#pragma unroll
        for (int v = 0; v < 8; v++) {
            int rib = wrow + mb + v;
            if (row0 + rib < cnt) {
                int   t = toks[rib];
                float w = rw[rib];
                atomicAdd(&out[(size_t)t * D_ + col], w * (cf[v] + bov));
            }
        }
    }
}

// ---------------- host launch ----------------
extern "C" void kernel_launch(void* const* d_in, const int* in_sizes, int n_in,
                              void* d_out, int out_size, void* d_ws, size_t ws_size,
                              hipStream_t stream) {
    const float* residual = (const float*)d_in[0];
    const float* W_router = (const float*)d_in[1];
    const float* W_gate   = (const float*)d_in[2];
    const float* b_gate   = (const float*)d_in[3];
    const float* W_up     = (const float*)d_in[4];
    const float* b_up     = (const float*)d_in[5];
    const float* W_out    = (const float*)d_in[6];
    const float* b_out    = (const float*)d_in[7];
    float* out = (float*)d_out;

    const size_t nW = (size_t)E_ * D_ * M_;        // 16,777,216
    const size_t nX = (size_t)TOKENS * D_;         // 4,194,304
    const size_t nH = (size_t)TOKENS * 2 * M_;     // 16,777,216 (top-2 assignments)

    char* w = (char*)d_ws;
    size_t off = 0;
    bf16* Wgt = (bf16*)(w + off); off += nW * sizeof(bf16);   // [E][M][D]
    bf16* Wut = (bf16*)(w + off); off += nW * sizeof(bf16);   // [E][M][D]
    bf16* Wot = (bf16*)(w + off); off += nW * sizeof(bf16);   // [E][D][M]
    bf16* Xb  = (bf16*)(w + off); off += nX * sizeof(bf16);
    bf16* H   = (bf16*)(w + off); off += nH * sizeof(bf16);
    int*   counts   = (int*)(w + off);   off += 256;
    int*   offsets  = (int*)(w + off);   off += 256;
    int*   list_tok = (int*)(w + off);   off += (size_t)E_ * TOKENS * sizeof(int);
    float* list_w   = (float*)(w + off); off += (size_t)E_ * TOKENS * sizeof(float);

    hipMemsetAsync(counts, 0, E_ * sizeof(int), stream);
    hipMemsetAsync(d_out, 0, (size_t)out_size * sizeof(float), stream);

    const int TB = 256;
    // pre-transpose + cast weights: gate/up [D][M]->[M][D], out [M][D]->[D][M]
    transpose_cast_kernel<<<dim3(M_ / 32, D_ / 32, E_), TB, 0, stream>>>(W_gate, Wgt, D_, M_);
    transpose_cast_kernel<<<dim3(M_ / 32, D_ / 32, E_), TB, 0, stream>>>(W_up,   Wut, D_, M_);
    transpose_cast_kernel<<<dim3(D_ / 32, M_ / 32, E_), TB, 0, stream>>>(W_out,  Wot, M_, D_);
    cast_f32_bf16_x4<<<(int)(nX / 4 / TB), TB, 0, stream>>>(residual, Xb, (int)(nX / 4));

    router_kernel<<<TOKENS, 32, 0, stream>>>(residual, W_router, counts, list_tok, list_w);
    scan_kernel<<<1, 32, 0, stream>>>(counts, offsets);

    gateup_kernel<<<dim3(M_ / 64, TOKENS / 64, E_), TB, 0, stream>>>(
        Xb, Wgt, Wut, b_gate, b_up, counts, offsets, list_tok, H);
    out_kernel<<<dim3(D_ / 64, TOKENS / 64, E_), TB, 0, stream>>>(
        H, Wot, b_out, counts, offsets, list_tok, list_w, out);
}
